// DPOTNet3D_14259291423360
// MI455X (gfx1250) — matrix operand order repaired and checked by reading, not compiled, not run on hardware
//
#include <hip/hip_runtime.h>
#include <math.h>

typedef __attribute__((ext_vector_type(16))) _Float16 v16h;
typedef __attribute__((ext_vector_type(8)))  float    v8f;

// Problem constants
constexpr int NBk = 8;      // blocks
constexpr int Cc  = 128;    // channels = NB*BS
constexpr int Bb  = 4, Hh = 64, Ww = 64, Ll = 32;
constexpr int MH = 32, MW = 32, MT = 8;          // kept modes
constexpr long NTOK = (long)Bb * MH * MW * MT;   // 32768 mode positions

// Packed-weight geometry: per block n, 16 B-tiles, each 32 lanes x 16 halves
//   tiles 0..3 : ain  dw-lo, dw-hi, fw-lo, fw-hi
//   tiles 4..7 : amid dw-lo, dw-hi, fw-lo, fw-hi
//   tiles 8..11: aout dw-lo, dw-hi, fw-lo, fw-hi
//   tile 12: [w1_0 ; -w1_1]   tile 13: [w1_1 ; w1_0]
//   tile 14: [w2_0 ; -w2_1]   tile 15: [w2_1 ; w2_0]
// Bias pack per n (256 f32): db0,fb0,db1,fb1,db2,fb2 (32 each), b1r,b1i,b2r,b2i (16 each)

__device__ __forceinline__ float gelu_f(float x) {
  return 0.5f * x * (1.0f + erff(x * 0.70710678118654752f));
}

// ---------------------------------------------------------------------------
// Weight prepack: global f32 -> WMMA B-layout f16 (done once, tiny)
// B 32x16 f16 layout: lane -> col = lane&15 ; elem e -> k = (lane>=16?16:0)+e
// ---------------------------------------------------------------------------
__global__ __launch_bounds__(256) void kern_pack_w(
    const float* __restrict__ w1, const float* __restrict__ w2,
    const float* __restrict__ a0dw, const float* __restrict__ a0fw,
    const float* __restrict__ a1dw, const float* __restrict__ a1fw,
    const float* __restrict__ a2dw, const float* __restrict__ a2fw,
    _Float16* __restrict__ gW)
{
  int id = (int)(blockIdx.x * 256 + threadIdx.x);   // one (n,tile,lane) each
  if (id >= NBk * 16 * 32) return;
  int lane = id & 31;
  int tile = (id >> 5) & 15;
  int n    = id >> 9;
  int ncol  = lane & 15;
  int kbase = (lane >> 4) << 4;
  _Float16* dst = gW + (size_t)id * 16;
#pragma unroll
  for (int e = 0; e < 16; ++e) {
    int k = kbase + e;
    float v = 0.0f;
    if (tile < 12) {
      int a = tile >> 2, sub = tile & 3;
      const float* dw = (a == 0 ? a0dw : (a == 1 ? a1dw : a2dw)) + (size_t)n * 512;
      const float* fw = (a == 0 ? a0fw : (a == 1 ? a1fw : a2fw)) + (size_t)n * 1024;
      if (sub == 0)      v = (k < 16) ? dw[k * 32 + ncol]      : 0.0f;
      else if (sub == 1) v = (k < 16) ? dw[k * 32 + 16 + ncol] : 0.0f;
      else if (sub == 2) v = fw[k * 32 + ncol];
      else               v = fw[k * 32 + 16 + ncol];
    } else {
      const float* wA = (tile < 14 ? w1 : w2) + (size_t)n * 256;         // w[0][n]
      const float* wB = (tile < 14 ? w1 : w2) + (size_t)(NBk + n) * 256; // w[1][n]
      if ((tile & 1) == 0) v = (k < 16) ? wA[k * 16 + ncol] : -wB[(k - 16) * 16 + ncol];
      else                 v = (k < 16) ? wB[k * 16 + ncol] :  wA[(k - 16) * 16 + ncol];
    }
    dst[e] = (_Float16)v;
  }
}

__global__ __launch_bounds__(256) void kern_pack_b(
    const float* __restrict__ b1, const float* __restrict__ b2,
    const float* __restrict__ a0db, const float* __restrict__ a0fb,
    const float* __restrict__ a1db, const float* __restrict__ a1fb,
    const float* __restrict__ a2db, const float* __restrict__ a2fb,
    float* __restrict__ gB)
{
  int id = (int)(blockIdx.x * 256 + threadIdx.x);   // 0..2047
  if (id >= NBk * 256) return;
  int n = id >> 8, j = id & 255;
  float v;
  if      (j <  32) v = a0db[n * 32 + j];
  else if (j <  64) v = a0fb[n * 32 + (j - 32)];
  else if (j <  96) v = a1db[n * 32 + (j - 64)];
  else if (j < 128) v = a1fb[n * 32 + (j - 96)];
  else if (j < 160) v = a2db[n * 32 + (j - 128)];
  else if (j < 192) v = a2fb[n * 32 + (j - 160)];
  else if (j < 208) v = b1[n * 16 + (j - 192)];            // b1[0][n]
  else if (j < 224) v = b1[(NBk + n) * 16 + (j - 208)];    // b1[1][n]
  else if (j < 240) v = b2[n * 16 + (j - 224)];            // b2[0][n]
  else              v = b2[(NBk + n) * 16 + (j - 240)];    // b2[1][n]
  gB[id] = v;
}

// ---------------------------------------------------------------------------
// WMMA tile helpers (wave32; C/D layout per CDNA5 ISA 7.12.2; A/B use this code's contiguous k order)
//   A 16x32 f16 : lane -> row = lane&15 ; elem e -> k = (lane>=16?16:0) + e
//   C/D 16x16 f32: elem i -> row = i + (lane>=16?8:0), col = lane&15
// B-tiles come prepacked from LDS: one 32-byte v16h per lane.
// ---------------------------------------------------------------------------
__device__ __forceinline__ v16h ldsB(const _Float16* sW, int tileIdx) {
  const int lane = (int)(threadIdx.x & 31u);
  return ((const v16h*)sW)[tileIdx * 32 + lane];
}

__device__ __forceinline__ void adapter_tile(
    int tbase, const _Float16* sW,
    const float* __restrict__ db, const float* __restrict__ fb,
    const float* sIn /*16x16*/, float* sH /*16x32*/, float* sOut /*16x16*/)
{
  const int lane = (int)(threadIdx.x & 31u);
  const int row  = lane & 15;
  const int koff = (lane >> 4) << 3;
  const int ncol = lane & 15;
  const int moff = (lane >> 4) << 3;

  // h = gelu(x @ dw + db) ; K padded 16->32 (pad zeros match packed zeros in B)
  v16h a;
#pragma unroll
  for (int e = 0; e < 16; ++e) {
    int k = (koff << 1) + e;
    a[e] = (k < 16) ? (_Float16)sIn[row * 16 + k] : (_Float16)0.0f;
  }
  v8f h0 = {}, h1 = {};
  h0 = __builtin_amdgcn_wmma_f32_16x16x32_f16(false, a, false, ldsB(sW, tbase + 0), (short)0, h0, false, false);
  h1 = __builtin_amdgcn_wmma_f32_16x16x32_f16(false, a, false, ldsB(sW, tbase + 1), (short)0, h1, false, false);
#pragma unroll
  for (int i = 0; i < 8; ++i) {
    int m = moff + i;
    sH[m * 32 + ncol]      = gelu_f(h0[i] + db[ncol]);
    sH[m * 32 + 16 + ncol] = gelu_f(h1[i] + db[16 + ncol]);
  }
  __syncthreads();

  // film = h @ fw + fb ; K=32 exact
  v16h a2;
#pragma unroll
  for (int e = 0; e < 16; ++e) {
    int k = (koff << 1) + e;
    a2[e] = (_Float16)sH[row * 32 + k];
  }
  v8f f0 = {}, f1 = {};
  f0 = __builtin_amdgcn_wmma_f32_16x16x32_f16(false, a2, false, ldsB(sW, tbase + 2), (short)0, f0, false, false);
  f1 = __builtin_amdgcn_wmma_f32_16x16x32_f16(false, a2, false, ldsB(sW, tbase + 3), (short)0, f1, false, false);
  // out = x * (1 + gamma) + beta
#pragma unroll
  for (int i = 0; i < 8; ++i) {
    int m = moff + i;
    float xv    = sIn[m * 16 + ncol];
    float gamma = f0[i] + fb[ncol];
    float beta  = f1[i] + fb[16 + ncol];
    sOut[m * 16 + ncol] = xv * (1.0f + gamma) + beta;
  }
  __syncthreads();
}

// Complex layer: A=[re‖im] (K=32), B prepacked as [[w0],[-w1]] / [[w1],[w0]]
__device__ __forceinline__ void clayer(
    int tcr, int tci, const _Float16* sW,
    const float* __restrict__ b0, const float* __restrict__ b1v,
    const float* sRin, const float* sIin,
    float* sRout, float* sIout, bool do_gelu)
{
  const int lane = (int)(threadIdx.x & 31u);
  const int row  = lane & 15;
  const int koff = (lane >> 4) << 3;
  const int ncol = lane & 15;
  const int moff = (lane >> 4) << 3;

  v16h a;
#pragma unroll
  for (int e = 0; e < 16; ++e) {
    int k = (koff << 1) + e;
    a[e] = (k < 16) ? (_Float16)sRin[row * 16 + k]
                    : (_Float16)sIin[row * 16 + (k - 16)];
  }
  v8f cr = {}, ci = {};
  cr = __builtin_amdgcn_wmma_f32_16x16x32_f16(false, a, false, ldsB(sW, tcr), (short)0, cr, false, false);
  ci = __builtin_amdgcn_wmma_f32_16x16x32_f16(false, a, false, ldsB(sW, tci), (short)0, ci, false, false);
#pragma unroll
  for (int i = 0; i < 8; ++i) {
    int m = moff + i;
    float vr = cr[i] + b0[ncol];
    float vi = ci[i] + b1v[ncol];
    if (do_gelu) { vr = gelu_f(vr); vi = gelu_f(vi); }
    sRout[m * 16 + ncol] = vr;
    sIout[m * 16 + ncol] = vi;
  }
  __syncthreads();
}

// ---------------------------------------------------------------------------
// Spectral kernel: grid (64, 8), 256 threads = 8 waves. Weights+biases staged
// in LDS once per workgroup; each wave processes 4 token tiles of 16 tokens.
// ---------------------------------------------------------------------------
__global__ __launch_bounds__(256) void kern_spectral(
    const float* __restrict__ Xr, const float* __restrict__ Xi,
    float* __restrict__ Yr, float* __restrict__ Yi,
    const _Float16* __restrict__ gW, const float* __restrict__ gB)
{
  __shared__ __align__(32) _Float16 sW[16 * 512];   // 16 KB packed B-tiles
  __shared__ float sB[256];                          // 1 KB biases
  __shared__ float sT[8 * 1536];                     // per-wave scratch (48 KB)

  const int tid  = (int)threadIdx.x;
  const int lane = tid & 31;
  const int wid  = tid >> 5;
  const int n    = (int)blockIdx.y;

  // Cooperative stage of this block's weights/biases
  {
    const uint4* src = (const uint4*)(gW + (size_t)n * 8192);
    uint4* dst = (uint4*)sW;
#pragma unroll
    for (int i = tid; i < 1024; i += 256) dst[i] = src[i];
    if (tid < 256) sB[tid] = gB[n * 256 + tid];
  }
  __syncthreads();

  float* my  = sT + wid * 1536;
  float* sXr = my;         // 16x16
  float* sXi = my + 256;   // 16x16
  float* sCr = my + 512;   // 16x16
  float* sCi = my + 768;   // 16x16
  float* sH  = my + 1024;  // 16x32

  const float* db0 = sB;       const float* fb0 = sB + 32;
  const float* db1 = sB + 64;  const float* fb1 = sB + 96;
  const float* db2 = sB + 128; const float* fb2 = sB + 160;
  const float* b1r = sB + 192; const float* b1i = sB + 208;
  const float* b2r = sB + 224; const float* b2i = sB + 240;

  for (int t = 0; t < 4; ++t) {
    const long tile = ((long)blockIdx.x * 8 + wid) * 4 + t;
    const long p0   = tile * 16;

#pragma unroll
    for (int i = 0; i < 8; ++i) {
      int idx = i * 32 + lane;
      int m = idx >> 4, col = idx & 15;
      long g = (p0 + m) * Cc + (long)n * 16 + col;
      sXr[m * 16 + col] = Xr[g];
      sXi[m * 16 + col] = Xi[g];
    }
    __syncthreads();

    adapter_tile(0, sW, db0, fb0, sXr, sH, sCr);             // ain (real)
    adapter_tile(0, sW, db0, fb0, sXi, sH, sCi);             // ain (imag)
    clayer(12, 13, sW, b1r, b1i, sCr, sCi, sXr, sXi, true);  // layer1 + gelu
    adapter_tile(4, sW, db1, fb1, sXr, sH, sCr);             // amid
    adapter_tile(4, sW, db1, fb1, sXi, sH, sCi);
    clayer(14, 15, sW, b2r, b2i, sCr, sCi, sXr, sXi, false); // layer2
    adapter_tile(8, sW, db2, fb2, sXr, sH, sCr);             // aout
    adapter_tile(8, sW, db2, fb2, sXi, sH, sCi);

#pragma unroll
    for (int i = 0; i < 8; ++i) {
      int idx = i * 32 + lane;
      int m = idx >> 4, col = idx & 15;
      long g = (p0 + m) * Cc + (long)n * 16 + col;
      Yr[g] = sCr[m * 16 + col];
      Yi[g] = sCi[m * 16 + col];
    }
    __syncthreads();
  }
}

// ---------------------------------------------------------------------------
// Truncated DFT stages (only kept modes computed; intermediates are L2-resident)
// ---------------------------------------------------------------------------

// Stage 1: real DFT along L: (B,H,W,32,C) -> (B,H,W,8,C). scale 1/sqrt(32)
__global__ __launch_bounds__(256) void kern_dft_l(
    const float* __restrict__ x, float* __restrict__ outR, float* __restrict__ outI)
{
  __shared__ float c32[32], s32[32];
  if (threadIdx.x < 32) {
    float ang = 6.28318530717958647692f * (float)threadIdx.x / 32.0f;
    c32[threadIdx.x] = cosf(ang);
    s32[threadIdx.x] = sinf(ang);
  }
  __syncthreads();
  long idx = (long)blockIdx.x * blockDim.x + threadIdx.x;
  if (idx >= (long)Bb * Hh * Ww * Cc) return;
  int c = (int)(idx & (Cc - 1));
  long bhw = idx >> 7;
  const float* xp = x + bhw * ((long)Ll * Cc) + c;
  float xv[32];
#pragma unroll
  for (int l = 0; l < 32; ++l) xv[l] = xp[(long)l * Cc];
  for (int k = 0; k < MT; ++k) {
    float ar = 0.0f, ai = 0.0f;
#pragma unroll
    for (int l = 0; l < 32; ++l) {
      int j = (k * l) & 31;
      ar += xv[l] * c32[j];
      ai -= xv[l] * s32[j];
    }
    long o = (bhw * MT + k) * Cc + c;
    outR[o] = ar * 0.17677669529663688f;
    outI[o] = ai * 0.17677669529663688f;
  }
}

// Stage 2: DFT along W (64 -> 32 modes). scale 1/8
__global__ __launch_bounds__(256) void kern_dft_w(
    const float* __restrict__ inR, const float* __restrict__ inI,
    float* __restrict__ outR, float* __restrict__ outI)
{
  __shared__ float c64[64], s64[64];
  if (threadIdx.x < 64) {
    float ang = 6.28318530717958647692f * (float)threadIdx.x / 64.0f;
    c64[threadIdx.x] = cosf(ang);
    s64[threadIdx.x] = sinf(ang);
  }
  __syncthreads();
  long idx = (long)blockIdx.x * blockDim.x + threadIdx.x;
  if (idx >= (long)Bb * Hh * MW * MT * Cc) return;
  int c  = (int)(idx & (Cc - 1));
  long t = idx >> 7;
  int kl = (int)(t & 7);  t >>= 3;
  int kw = (int)(t & 31); t >>= 5;
  int h  = (int)(t & 63);
  int b  = (int)(t >> 6);
  long base = ((((long)b * Hh + h) * Ww) * MT + kl) * Cc + c;
  const long wstride = (long)MT * Cc;
  float ar = 0.0f, ai = 0.0f;
  for (int w = 0; w < 64; ++w) {
    int j = (kw * w) & 63;
    float xr = inR[base + (long)w * wstride];
    float xi = inI[base + (long)w * wstride];
    float cc = c64[j], ss = s64[j];
    ar += xr * cc + xi * ss;   // * e^{-i theta}
    ai += xi * cc - xr * ss;
  }
  long o = ((((long)b * Hh + h) * MW + kw) * MT + kl) * Cc + c;
  outR[o] = ar * 0.125f;
  outI[o] = ai * 0.125f;
}

// Stage 3: DFT along H (64 -> 32 modes). scale 1/8
__global__ __launch_bounds__(256) void kern_dft_h(
    const float* __restrict__ inR, const float* __restrict__ inI,
    float* __restrict__ outR, float* __restrict__ outI)
{
  __shared__ float c64[64], s64[64];
  if (threadIdx.x < 64) {
    float ang = 6.28318530717958647692f * (float)threadIdx.x / 64.0f;
    c64[threadIdx.x] = cosf(ang);
    s64[threadIdx.x] = sinf(ang);
  }
  __syncthreads();
  long idx = (long)blockIdx.x * blockDim.x + threadIdx.x;
  if (idx >= (long)Bb * MH * MW * MT * Cc) return;
  int c  = (int)(idx & (Cc - 1));
  long t = idx >> 7;
  int kl = (int)(t & 7);  t >>= 3;
  int kw = (int)(t & 31); t >>= 5;
  int kh = (int)(t & 31);
  int b  = (int)(t >> 5);
  long base = ((((long)b * Hh) * MW + kw) * MT + kl) * Cc + c;
  const long hstride = (long)MW * MT * Cc;
  float ar = 0.0f, ai = 0.0f;
  for (int h = 0; h < 64; ++h) {
    int j = (kh * h) & 63;
    float xr = inR[base + (long)h * hstride];
    float xi = inI[base + (long)h * hstride];
    float cc = c64[j], ss = s64[j];
    ar += xr * cc + xi * ss;
    ai += xi * cc - xr * ss;
  }
  long o = ((((long)b * MH + kh) * MW + kw) * MT + kl) * Cc + c;
  outR[o] = ar * 0.125f;
  outI[o] = ai * 0.125f;
}

// Stage 5: inverse DFT along H (32 modes -> 64). scale 1/8
__global__ __launch_bounds__(256) void kern_idft_h(
    const float* __restrict__ inR, const float* __restrict__ inI,
    float* __restrict__ outR, float* __restrict__ outI)
{
  __shared__ float c64[64], s64[64];
  if (threadIdx.x < 64) {
    float ang = 6.28318530717958647692f * (float)threadIdx.x / 64.0f;
    c64[threadIdx.x] = cosf(ang);
    s64[threadIdx.x] = sinf(ang);
  }
  __syncthreads();
  long idx = (long)blockIdx.x * blockDim.x + threadIdx.x;
  if (idx >= (long)Bb * Hh * MW * MT * Cc) return;
  int c  = (int)(idx & (Cc - 1));
  long t = idx >> 7;
  int kl = (int)(t & 7);  t >>= 3;
  int kw = (int)(t & 31); t >>= 5;
  int h  = (int)(t & 63);
  int b  = (int)(t >> 6);
  long base = ((((long)b * MH) * MW + kw) * MT + kl) * Cc + c;
  const long khstride = (long)MW * MT * Cc;
  float ar = 0.0f, ai = 0.0f;
  for (int kh = 0; kh < 32; ++kh) {
    int j = (kh * h) & 63;
    float xr = inR[base + (long)kh * khstride];
    float xi = inI[base + (long)kh * khstride];
    float cc = c64[j], ss = s64[j];
    ar += xr * cc - xi * ss;   // * e^{+i theta}
    ai += xi * cc + xr * ss;
  }
  long o = ((((long)b * Hh + h) * MW + kw) * MT + kl) * Cc + c;
  outR[o] = ar * 0.125f;
  outI[o] = ai * 0.125f;
}

// Stage 6: inverse DFT along W (32 modes -> 64). scale 1/8
__global__ __launch_bounds__(256) void kern_idft_w(
    const float* __restrict__ inR, const float* __restrict__ inI,
    float* __restrict__ outR, float* __restrict__ outI)
{
  __shared__ float c64[64], s64[64];
  if (threadIdx.x < 64) {
    float ang = 6.28318530717958647692f * (float)threadIdx.x / 64.0f;
    c64[threadIdx.x] = cosf(ang);
    s64[threadIdx.x] = sinf(ang);
  }
  __syncthreads();
  long idx = (long)blockIdx.x * blockDim.x + threadIdx.x;
  if (idx >= (long)Bb * Hh * Ww * MT * Cc) return;
  int c  = (int)(idx & (Cc - 1));
  long t = idx >> 7;
  int kl = (int)(t & 7);  t >>= 3;
  int w  = (int)(t & 63); t >>= 6;
  int h  = (int)(t & 63);
  int b  = (int)(t >> 6);
  long base = ((((long)b * Hh + h) * MW) * MT + kl) * Cc + c;
  const long kwstride = (long)MT * Cc;
  float ar = 0.0f, ai = 0.0f;
  for (int kw = 0; kw < 32; ++kw) {
    int j = (kw * w) & 63;
    float xr = inR[base + (long)kw * kwstride];
    float xi = inI[base + (long)kw * kwstride];
    float cc = c64[j], ss = s64[j];
    ar += xr * cc - xi * ss;
    ai += xi * cc + xr * ss;
  }
  long o = ((((long)b * Hh + h) * Ww + w) * MT + kl) * Cc + c;
  outR[o] = ar * 0.125f;
  outI[o] = ai * 0.125f;
}

// Stage 7: inverse real DFT along L (8 bins -> 32) + residual, streaming store.
__global__ __launch_bounds__(256) void kern_irfft_l(
    const float* __restrict__ inR, const float* __restrict__ inI,
    const float* __restrict__ x, float* __restrict__ y)
{
  __shared__ float c32[32], s32[32];
  if (threadIdx.x < 32) {
    float ang = 6.28318530717958647692f * (float)threadIdx.x / 32.0f;
    c32[threadIdx.x] = cosf(ang);
    s32[threadIdx.x] = sinf(ang);
  }
  __syncthreads();
  long idx = (long)blockIdx.x * blockDim.x + threadIdx.x;
  if (idx >= (long)Bb * Hh * Ww * Cc) return;
  int c = (int)(idx & (Cc - 1));
  long bhw = idx >> 7;
  float vr[8], vi[8];
#pragma unroll
  for (int k = 0; k < 8; ++k) {
    long g = (bhw * MT + k) * Cc + c;
    vr[k] = inR[g];
    vi[k] = inI[g];
  }
  for (int l = 0; l < 32; ++l) {
    float acc = vr[0];
#pragma unroll
    for (int k = 1; k < 8; ++k) {
      int j = (k * l) & 31;
      acc += 2.0f * (vr[k] * c32[j] - vi[k] * s32[j]);
    }
    long o = (bhw * Ll + l) * Cc + c;
    float val = __builtin_nontemporal_load(&x[o]) + acc * 0.17677669529663688f;
    __builtin_nontemporal_store(val, &y[o]);
  }
}

// ---------------------------------------------------------------------------
// Host-side launch
// ---------------------------------------------------------------------------
extern "C" void kernel_launch(void* const* d_in, const int* in_sizes, int n_in,
                              void* d_out, int out_size, void* d_ws, size_t ws_size,
                              hipStream_t stream)
{
  (void)in_sizes; (void)n_in; (void)out_size; (void)ws_size;
  const float* x    = (const float*)d_in[0];
  const float* w1   = (const float*)d_in[1];
  const float* b1   = (const float*)d_in[2];
  const float* w2   = (const float*)d_in[3];
  const float* b2   = (const float*)d_in[4];
  const float* a0dw = (const float*)d_in[5];
  const float* a0db = (const float*)d_in[6];
  const float* a0fw = (const float*)d_in[7];
  const float* a0fb = (const float*)d_in[8];
  const float* a1dw = (const float*)d_in[9];
  const float* a1db = (const float*)d_in[10];
  const float* a1fw = (const float*)d_in[11];
  const float* a1fb = (const float*)d_in[12];
  const float* a2dw = (const float*)d_in[13];
  const float* a2db = (const float*)d_in[14];
  const float* a2fw = (const float*)d_in[15];
  const float* a2fb = (const float*)d_in[16];
  float* y = (float*)d_out;

  // Workspace layout (floats): DFT buffers (256 MiB) + packed weights/biases
  float* ws = (float*)d_ws;
  float* Ar = ws;                    // 16,777,216  (B,H,W,8,C)
  float* Ai = ws + 16777216L;
  float* Br = ws + 33554432L;        //  8,388,608  (B,H,32,8,C)
  float* Bi = ws + 41943040L;
  float* Cr = ws + 50331648L;        //  4,194,304  (B,32,32,8,C)
  float* Ci = ws + 54525952L;
  float* Dr = ws + 58720256L;        //  4,194,304  spectral output
  float* Di = ws + 62914560L;
  _Float16* gW = (_Float16*)(ws + 67108864L);   // 65,536 halves (128 KB)
  float*    gB = ws + 67108864L + 32768L;       //  2,048 floats (8 KB)

  // Weight/bias prepack (independent of DFT stages)
  kern_pack_w<<<16, 256, 0, stream>>>(w1, w2, a0dw, a0fw, a1dw, a1fw, a2dw, a2fw, gW);
  kern_pack_b<<< 8, 256, 0, stream>>>(b1, b2, a0db, a0fb, a1db, a1fb, a2db, a2fb, gB);

  kern_dft_l <<<  8192, 256, 0, stream>>>(x, Ar, Ai);          // x -> A
  kern_dft_w <<< 32768, 256, 0, stream>>>(Ar, Ai, Br, Bi);     // A -> B
  kern_dft_h <<< 16384, 256, 0, stream>>>(Br, Bi, Cr, Ci);     // B -> C

  dim3 gs(64, NBk, 1);  // 64 WGs x 8 waves x 4 tiles = 2048 tiles per block-n
  kern_spectral<<<gs, 256, 0, stream>>>(Cr, Ci, Dr, Di, gW, gB);

  kern_idft_h<<< 32768, 256, 0, stream>>>(Dr, Di, Br, Bi);     // D -> B (reuse)
  kern_idft_w<<< 65536, 256, 0, stream>>>(Br, Bi, Ar, Ai);     // B -> A (reuse)
  kern_irfft_l<<<  8192, 256, 0, stream>>>(Ar, Ai, x, y);      // A + x -> y
}
